// LSAAttention3D_62474594287826
// MI455X (gfx1250) — matrix operand order repaired and checked by reading, not compiled, not run on hardware
//
#include <hip/hip_runtime.h>

typedef __bf16 bf16_t;
typedef bf16_t v16bf __attribute__((ext_vector_type(16)));
typedef bf16_t v8bf  __attribute__((ext_vector_type(8)));
typedef bf16_t v4bf  __attribute__((ext_vector_type(4)));
typedef float  v8f   __attribute__((ext_vector_type(8)));

#define SX 104            // row stride (halfs) for 96-col matrices (16B-aligned rows)
#define SA 232            // row stride (halfs) for 224-col matrices (attn, V^T)
#define NWAVES 7
#define NTHREADS 224
#define WIN 216
#define MPAD 224
#define CDIM 96
#define GROUPS 1728
#define NTOT 373248
#define QKSCALE 0.288675134594813f  // (96/8)^-0.5

// shared-memory pool offsets (bf16 elements)
#define SZ_Q  (MPAD * SX)                 // 23296
#define SZ_VT (CDIM * SA)                 // 22272
#define SZ_W  (CDIM * SX)                 // 9984
#define OFF_Q  0
#define OFF_K  (OFF_Q + SZ_Q)
#define OFF_VT (OFF_K + SZ_Q)
#define OFF_WP (OFF_VT + SZ_VT)
#define OFF_SH (OFF_WP + SZ_W)
#define SZ_SH  (SZ_Q + 3 * SZ_W)          // X+Wq+Wk+Wv = 53248 >= attn (224*232=51968)
#define SMEM_TOTAL (OFF_SH + SZ_SH)       // 132096 halfs = 258 KB

__device__ __forceinline__ v8f wmma_bf16(v16bf a, v16bf b, v8f c) {
  return __builtin_amdgcn_wmma_f32_16x16x32_bf16(false, a, false, b, (short)0, c,
                                                 false, false);
}

// A-matrix 16x32 bf16 fragment from a row-major LDS matrix.
__device__ __forceinline__ v16bf load_afrag(const bf16_t* M, int stride, int m0,
                                            int k0, int lane) {
  const int row = m0 + (lane & 15);
  const int hi  = lane >> 4;
  const bf16_t* p = M + row * stride + k0 + hi * 8;
  v8bf lo = *(const v8bf*)p;
  v8bf hp = *(const v8bf*)(p + 16);
  v16bf a;
#pragma unroll
  for (int i = 0; i < 8; ++i) { a[i] = lo[i]; a[i + 8] = hp[i]; }
  return a;
}

// B-matrix 32x16 bf16 fragment where B[k][n] = SRC[n][k], SRC row-major.
__device__ __forceinline__ v16bf load_bfrag(const bf16_t* SRC, int stride, int n0,
                                            int k0, int lane) {
  const int col = n0 + (lane & 15);
  const int hi  = lane >> 4;
  const bf16_t* p = SRC + col * stride + k0 + hi * 8;
  v8bf lo = *(const v8bf*)p;
  v8bf hp = *(const v8bf*)(p + 16);
  v16bf b;
#pragma unroll
  for (int i = 0; i < 8; ++i) { b[i] = lo[i]; b[i + 8] = hp[i]; }
  return b;
}

__global__ __launch_bounds__(NTHREADS, 1)
void lsa3d_kernel(const float* __restrict__ x,
                  const float* __restrict__ wq, const float* __restrict__ wk,
                  const float* __restrict__ wv, const float* __restrict__ wp,
                  const float* __restrict__ bp, float* __restrict__ out) {
  __shared__ __align__(16) bf16_t smem[SMEM_TOTAL];
  __shared__ float sBias[CDIM];

  bf16_t* sQ    = smem + OFF_Q;
  bf16_t* sK    = smem + OFF_K;
  bf16_t* sVt   = smem + OFF_VT;          // V transposed: [d][m], stride SA
  bf16_t* sWp   = smem + OFF_WP;
  bf16_t* sX    = smem + OFF_SH;
  bf16_t* sWq   = sX + MPAD * SX;
  bf16_t* sWk   = sWq + CDIM * SX;
  bf16_t* sWv   = sWk + CDIM * SX;
  bf16_t* sAttn = smem + OFF_SH;          // aliases X/Wq/Wk/Wv after phase 1
  bf16_t* sO    = sQ;                     // aliases Q after phase 2

  const int tid  = threadIdx.x;
  // wave index is uniform within a wave: force it scalar so job indices,
  // branches and most address math become SALU, not per-lane VALU + exec masks.
  const int wave = __builtin_amdgcn_readfirstlane(tid >> 5);
  const int lane = tid & 31;

  const int bg = blockIdx.x;
  const int b  = bg / GROUPS;
  const int g  = bg % GROUPS;
  const float* xg   = x   + (size_t)b * CDIM * NTOT + (size_t)g * WIN * CDIM;
  float*       outg = out + (size_t)b * CDIM * NTOT + (size_t)g * WIN;

  // ---------------- Phase 0: stage X (bf16, pad rows zeroed) + weights + bias
  {
    const float4* xg4 = (const float4*)xg;
    for (int i = tid; i < (WIN * CDIM) / 4; i += NTHREADS) {  // 5184 float4
      float4 v = xg4[i];
      const int e = i * 4;
      const int r = e / CDIM, c = e % CDIM;  // c multiple of 4
      v4bf o;
      o[0] = (bf16_t)v.x; o[1] = (bf16_t)v.y; o[2] = (bf16_t)v.z; o[3] = (bf16_t)v.w;
      *(v4bf*)(sX + r * SX + c) = o;
    }
    if (tid < (8 * CDIM) / 4) {                               // zero pad rows
      const int e = tid * 4;
      const int r = WIN + e / CDIM, c = e % CDIM;
      v4bf z; z[0] = z[1] = z[2] = z[3] = (bf16_t)0.f;
      *(v4bf*)(sX + r * SX + c) = z;
    }
    const float4* wq4 = (const float4*)wq;
    const float4* wk4 = (const float4*)wk;
    const float4* wv4 = (const float4*)wv;
    const float4* wp4 = (const float4*)wp;
    for (int i = tid; i < (CDIM * CDIM) / 4; i += NTHREADS) {  // 2304 float4 each
      const int e = i * 4;
      const int r = e / CDIM, c = e % CDIM;
      float4 a = wq4[i], d = wk4[i], f = wv4[i], h = wp4[i];
      v4bf oa, od, of, oh;
      oa[0] = (bf16_t)a.x; oa[1] = (bf16_t)a.y; oa[2] = (bf16_t)a.z; oa[3] = (bf16_t)a.w;
      od[0] = (bf16_t)d.x; od[1] = (bf16_t)d.y; od[2] = (bf16_t)d.z; od[3] = (bf16_t)d.w;
      of[0] = (bf16_t)f.x; of[1] = (bf16_t)f.y; of[2] = (bf16_t)f.z; of[3] = (bf16_t)f.w;
      oh[0] = (bf16_t)h.x; oh[1] = (bf16_t)h.y; oh[2] = (bf16_t)h.z; oh[3] = (bf16_t)h.w;
      *(v4bf*)(sWq + r * SX + c) = oa;
      *(v4bf*)(sWk + r * SX + c) = od;
      *(v4bf*)(sWv + r * SX + c) = of;
      *(v4bf*)(sWp + r * SX + c) = oh;
    }
    if (tid < CDIM) sBias[tid] = bp[tid];
  }
  __syncthreads();

  // ---------------- Phase 1: Q = scale*(X Wq^T), K = X Wk^T, V^T from X Wv^T
  for (int ob = wave; ob < 18; ob += NWAVES) {  // 3 mats x 6 n-tiles (scalar)
    const int mat = ob / 6;
    const int nt  = ob % 6;
    const bf16_t* Ws = (mat == 0) ? sWq : (mat == 1) ? sWk : sWv;
    v16bf b0 = load_bfrag(Ws, SX, nt * 16, 0,  lane);
    v16bf b1 = load_bfrag(Ws, SX, nt * 16, 32, lane);
    v16bf b2 = load_bfrag(Ws, SX, nt * 16, 64, lane);
    for (int mt = 0; mt < 14; ++mt) {
      v16bf a0 = load_afrag(sX, SX, mt * 16, 0,  lane);
      v16bf a1 = load_afrag(sX, SX, mt * 16, 32, lane);
      v16bf a2 = load_afrag(sX, SX, mt * 16, 64, lane);
      v8f acc = {0.f, 0.f, 0.f, 0.f, 0.f, 0.f, 0.f, 0.f};
      acc = wmma_bf16(a0, b0, acc);
      acc = wmma_bf16(a1, b1, acc);
      acc = wmma_bf16(a2, b2, acc);
      const int col   = nt * 16 + (lane & 15);
      const int rbase = mt * 16 + 8 * (lane >> 4);
      if (mat == 2) {                 // scalar branch: V^T, 8 contiguous m per lane
        v8bf o8;
#pragma unroll
        for (int i = 0; i < 8; ++i) o8[i] = (bf16_t)acc[i];
        *(v8bf*)(sVt + col * SA + rbase) = o8;
      } else {
        bf16_t* dst = (mat == 0) ? sQ : sK;
        const float sc = (mat == 0) ? QKSCALE : 1.0f;
#pragma unroll
        for (int i = 0; i < 8; ++i) dst[(rbase + i) * SX + col] = (bf16_t)(acc[i] * sc);
      }
    }
  }
  __syncthreads();

  // ---------------- Phase 2: scores S = Q K^T  (scale already in Q)
  for (int jt = wave; jt < 14; jt += NWAVES) {
    v16bf b0 = load_bfrag(sK, SX, jt * 16, 0,  lane);
    v16bf b1 = load_bfrag(sK, SX, jt * 16, 32, lane);
    v16bf b2 = load_bfrag(sK, SX, jt * 16, 64, lane);
    for (int mt = 0; mt < 14; ++mt) {
      v16bf a0 = load_afrag(sQ, SX, mt * 16, 0,  lane);
      v16bf a1 = load_afrag(sQ, SX, mt * 16, 32, lane);
      v16bf a2 = load_afrag(sQ, SX, mt * 16, 64, lane);
      v8f acc = {0.f, 0.f, 0.f, 0.f, 0.f, 0.f, 0.f, 0.f};
      acc = wmma_bf16(a0, b0, acc);
      acc = wmma_bf16(a1, b1, acc);
      acc = wmma_bf16(a2, b2, acc);
      const int col   = jt * 16 + (lane & 15);
      const int rbase = mt * 16 + 8 * (lane >> 4);
#pragma unroll
      for (int i = 0; i < 8; ++i) sAttn[(rbase + i) * SA + col] = (bf16_t)acc[i];
    }
  }
  __syncthreads();

  // ---------------- Phase 3: row softmax over the 216 valid columns (27 x v8bf)
  if (tid < WIN) {
    bf16_t* row = sAttn + tid * SA;
    float mx = -1e30f;
    for (int j = 0; j < 27; ++j) {
      v8bf v = *(const v8bf*)(row + j * 8);
#pragma unroll
      for (int i = 0; i < 8; ++i) mx = fmaxf(mx, (float)v[i]);
    }
    float sum = 0.f;
    for (int j = 0; j < 27; ++j) {
      v8bf v = *(const v8bf*)(row + j * 8);
      v8bf e8;
#pragma unroll
      for (int i = 0; i < 8; ++i) {
        float e = __expf((float)v[i] - mx);
        sum += e;
        e8[i] = (bf16_t)e;
      }
      *(v8bf*)(row + j * 8) = e8;
    }
    const float inv = 1.f / sum;
    for (int j = 0; j < 27; ++j) {
      v8bf v = *(const v8bf*)(row + j * 8);
#pragma unroll
      for (int i = 0; i < 8; ++i) v[i] = (bf16_t)((float)v[i] * inv);
      *(v8bf*)(row + j * 8) = v;
    }
    v8bf z;
#pragma unroll
    for (int i = 0; i < 8; ++i) z[i] = (bf16_t)0.f;
    *(v8bf*)(row + WIN) = z;            // kill padded K-columns 216..223
  }
  __syncthreads();

  // ---------------- Phase 4: O = softmax(S) · V   (K-dim = 224, 7 chunks)
  for (int job = wave; job < 84; job += NWAVES) {
    const int nt = job / 14;
    const int mt = job % 14;
    v16bf af[7], bfr[7];
#pragma unroll
    for (int kc = 0; kc < 7; ++kc) {
      af[kc]  = load_afrag(sAttn, SA, mt * 16, kc * 32, lane);
      bfr[kc] = load_bfrag(sVt,   SA, nt * 16, kc * 32, lane);  // Vt[d][m]
    }
    v8f acc = {0.f, 0.f, 0.f, 0.f, 0.f, 0.f, 0.f, 0.f};
#pragma unroll
    for (int kc = 0; kc < 7; ++kc) acc = wmma_bf16(af[kc], bfr[kc], acc);
    const int col   = nt * 16 + (lane & 15);
    const int rbase = mt * 16 + 8 * (lane >> 4);
#pragma unroll
    for (int i = 0; i < 8; ++i) sO[(rbase + i) * SX + col] = (bf16_t)acc[i];
  }
  __syncthreads();

  // ---------------- Phase 5: Y = O Wp^T + bp, store channel-major
  for (int job = wave; job < 84; job += NWAVES) {
    const int nt = job / 14;
    const int mt = job % 14;
    v16bf a0 = load_afrag(sO, SX, mt * 16, 0,  lane);
    v16bf a1 = load_afrag(sO, SX, mt * 16, 32, lane);
    v16bf a2 = load_afrag(sO, SX, mt * 16, 64, lane);
    v16bf b0 = load_bfrag(sWp, SX, nt * 16, 0,  lane);
    v16bf b1 = load_bfrag(sWp, SX, nt * 16, 32, lane);
    v16bf b2 = load_bfrag(sWp, SX, nt * 16, 64, lane);
    v8f acc = {0.f, 0.f, 0.f, 0.f, 0.f, 0.f, 0.f, 0.f};
    acc = wmma_bf16(a0, b0, acc);
    acc = wmma_bf16(a1, b1, acc);
    acc = wmma_bf16(a2, b2, acc);
    const int c    = nt * 16 + (lane & 15);
    const int tok0 = mt * 16 + 8 * (lane >> 4);
    if (tok0 < WIN) {                          // drop padded tokens 216..223
      const float bias = sBias[c];
      float* dst = outg + (size_t)c * NTOT + tok0;
      float4 vlo = {acc[0] + bias, acc[1] + bias, acc[2] + bias, acc[3] + bias};
      float4 vhi = {acc[4] + bias, acc[5] + bias, acc[6] + bias, acc[7] + bias};
      *(float4*)dst       = vlo;
      *(float4*)(dst + 4) = vhi;
    }
  }
}

extern "C" void kernel_launch(void* const* d_in, const int* in_sizes, int n_in,
                              void* d_out, int out_size, void* d_ws, size_t ws_size,
                              hipStream_t stream) {
  (void)in_sizes; (void)n_in; (void)out_size; (void)d_ws; (void)ws_size;
  const float* x  = (const float*)d_in[0];
  const float* wq = (const float*)d_in[1];
  const float* wk = (const float*)d_in[2];
  const float* wv = (const float*)d_in[3];
  const float* wp = (const float*)d_in[4];
  const float* bp = (const float*)d_in[5];
  float* out = (float*)d_out;

  dim3 grid(2 * GROUPS);      // 3456 windows
  dim3 block(NTHREADS);       // 7 waves
  lsa3d_kernel<<<grid, block, 0, stream>>>(x, wq, wk, wv, wp, bp, out);
}
